// LSTMCell_72121090835050
// MI455X (gfx1250) — compile-verified, matrix operand order checked by reading, not run on hardware
//
#include <hip/hip_runtime.h>

// ---------------------------------------------------------------------------
// LSTM cell, fused:  Z = [x|h] @ [Wx|Uh]^T  (bf16 WMMA, f32 accum) + gates
//   B=8192, D=1024, H=1024, gates (f,i,o,g) along 4*H output columns.
//
// Pass 1: pack_bf16  — convert x|h -> Ab (8192x2048 bf16) and Wx|Uh -> Wb
//         (4096x2048 bf16) in workspace (needs 48 MB of d_ws).
// Pass 2: lstm_cell_wmma — 128 rows x 64 h-cols x 4 gates per block,
//         8 wave32, double-buffered LDS fed by global_load_async_to_lds_b128
//         (ASYNCcnt), 16 v_wmma_f32_16x16x32_bf16 per wave per K-step,
//         fused sigmoid/tanh/c-update epilogue.
// ---------------------------------------------------------------------------

typedef __attribute__((ext_vector_type(16))) __bf16 v16bf;
typedef __attribute__((ext_vector_type(8)))  __bf16 v8bf;
typedef __attribute__((ext_vector_type(8)))  float  v8f;

#define NB 8192
#define ND 1024
#define NH 1024
#define KTOT 2048
#define BK 32
#define LDSW 40   // padded LDS row stride in halves (80B, 16B-aligned, conflict-free)

__device__ __forceinline__ v16bf cat8(v8bf lo, v8bf hi) {
  return __builtin_shufflevector(lo, hi, 0,1,2,3,4,5,6,7,8,9,10,11,12,13,14,15);
}

__device__ __forceinline__ float fast_sigmoid(float x) {
  return __builtin_amdgcn_rcpf(1.0f + __expf(-x));        // v_exp + v_rcp, branch-free
}
__device__ __forceinline__ float fast_tanh(float x) {
  return 1.0f - 2.0f * __builtin_amdgcn_rcpf(__expf(2.0f * x) + 1.0f);
}

// Async DMA: global -> LDS, 16 bytes per lane.  Generic pointers to __shared__
// carry the LDS byte offset in their low 32 bits (ISA 10.2 aperture mapping).
__device__ __forceinline__ void async_b128(const __bf16* gp, const __bf16* lp) {
  asm volatile("global_load_async_to_lds_b128 %0, %1, off"
               :: "v"((unsigned)(size_t)lp),
                  "v"((unsigned long long)(size_t)gp)
               : "memory");
}
__device__ __forceinline__ void wait_async0() {
  asm volatile("s_wait_asynccnt 0x0" ::: "memory");
}

// ---------------------------------------------------------------------------
// Pass 1: f32 -> bf16 packing.  One thread per 8 elements.
//   rows [0,8192)    : Ab[row][k] = k<1024 ? x[row][k] : hp[row][k-1024]
//   rows [8192,12288): Wb[r][k]   = k<1024 ? Wx.flat[r][k] : Uh.flat[r][k-1024]
// ---------------------------------------------------------------------------
__global__ __launch_bounds__(256)
void pack_bf16(const float* __restrict__ x,  const float* __restrict__ hp,
               const float* __restrict__ Wx, const float* __restrict__ Uh,
               __bf16* __restrict__ Ab, __bf16* __restrict__ Wb)
{
  const int idx = blockIdx.x * 256 + threadIdx.x;   // one 8-element group
  const int row = idx >> 8;                         // 256 groups per 2048-row
  const int kg  = (idx & 255) * 8;

  const float* src;
  __bf16* dst;
  if (row < NB) {
    src = (kg < ND) ? (x  + (size_t)row * ND + kg)
                    : (hp + (size_t)row * NH + (kg - ND));
    dst = Ab + (size_t)row * KTOT + kg;
  } else {
    const int r = row - NB;                         // 0..4095 = g*1024+h
    src = (kg < ND) ? (Wx + (size_t)r * ND + kg)
                    : (Uh + (size_t)r * NH + (kg - ND));
    dst = Wb + (size_t)r * KTOT + kg;
  }
  const float4 v0 = *(const float4*)(src);
  const float4 v1 = *(const float4*)(src + 4);
  v8bf pk;
  pk[0] = (__bf16)v0.x; pk[1] = (__bf16)v0.y;
  pk[2] = (__bf16)v0.z; pk[3] = (__bf16)v0.w;
  pk[4] = (__bf16)v1.x; pk[5] = (__bf16)v1.y;
  pk[6] = (__bf16)v1.z; pk[7] = (__bf16)v1.w;
  *(v8bf*)dst = pk;
}

// ---------------------------------------------------------------------------
// Pass 2: fused WMMA GEMM + gates.
// ---------------------------------------------------------------------------
__global__ __launch_bounds__(256)
void lstm_cell_wmma(const __bf16* __restrict__ Ab,  // (8192,2048)
                    const __bf16* __restrict__ Wb,  // (4096,2048)
                    const float*  __restrict__ cp,  // (B,H)
                    const float*  __restrict__ bW,  // (4,H)
                    const float*  __restrict__ bU,  // (4,H)
                    float* __restrict__ out)        // [h_t (B,H) ; c_t (B,H)]
{
  __shared__ __bf16 As[2][128][LDSW];      // batch-rows  x K-chunk, ping-pong
  __shared__ __bf16 Bs[2][4][64][LDSW];    // gate, h-col x K-chunk, ping-pong

  const int tid     = threadIdx.x;
  const int lane    = tid & 31;
  const int wid     = tid >> 5;            // 0..7
  const int waveM   = wid & 3;             // 4 sub-tiles of 32 rows
  const int waveN   = wid >> 2;            // 2 sub-tiles of 32 h-cols
  const int lm      = lane & 15;           // m (or n) within a 16x16 tile
  const int hk      = lane >> 4;           // K half selector

  const int rowBase = blockIdx.y * 128;    // batch rows
  const int colBase = blockIdx.x * 64;     // h columns

  v8f acc[4][2][2];
  #pragma unroll
  for (int g = 0; g < 4; ++g)
    #pragma unroll
    for (int ms = 0; ms < 2; ++ms)
      #pragma unroll
      for (int ns = 0; ns < 2; ++ns)
        acc[g][ms][ns] = (v8f){0.f,0.f,0.f,0.f,0.f,0.f,0.f,0.f};

  // ---- staging geometry: thread -> one b128 chunk, 6 chunks per thread ----
  const int sr = tid >> 2;                 // 0..63
  const int c8 = (tid & 3) * 8;            // half-column group {0,8,16,24}
  // A rows sr, sr+64;  B gate p in 0..3, row sr.
  const __bf16* pa[2];
  const __bf16* pb[4];
  #pragma unroll
  for (int p = 0; p < 2; ++p)
    pa[p] = Ab + (size_t)(rowBase + sr + 64 * p) * KTOT + c8;
  #pragma unroll
  for (int p = 0; p < 4; ++p)
    pb[p] = Wb + (size_t)(p * NH + colBase + sr) * KTOT + c8;

  auto issue_tile = [&](int buf) {
    #pragma unroll
    for (int p = 0; p < 2; ++p)
      async_b128(pa[p], &As[buf][sr + 64 * p][c8]);
    #pragma unroll
    for (int p = 0; p < 4; ++p)
      async_b128(pb[p], &Bs[buf][p][sr][c8]);
  };

  auto compute = [&](int buf) {
    // A fragments (ISA 16-bit A 16x32 layout): two ds_load_b128 each
    v16bf afrag[2];
    #pragma unroll
    for (int ms = 0; ms < 2; ++ms) {
      const __bf16* rp = &As[buf][waveM * 32 + ms * 16 + lm][0];
      afrag[ms] = cat8(*(const v8bf*)(rp + hk * 8),
                       *(const v8bf*)(rp + 16 + hk * 8));
    }
    #pragma unroll
    for (int g = 0; g < 4; ++g) {
      #pragma unroll
      for (int ns = 0; ns < 2; ++ns) {
        const __bf16* rp = &Bs[buf][g][waveN * 32 + ns * 16 + lm][0];
        const v16bf bfrag = cat8(*(const v8bf*)(rp + hk * 8),
                                 *(const v8bf*)(rp + 16 + hk * 8));
        #pragma unroll
        for (int ms = 0; ms < 2; ++ms) {
          acc[g][ms][ns] = __builtin_amdgcn_wmma_f32_16x16x32_bf16(
              false, afrag[ms], false, bfrag, (short)0, acc[g][ms][ns], false, false);
        }
      }
    }
  };

  // ---- software pipeline: DMA tile k+1 overlaps WMMAs on tile k ----
  issue_tile(0);
  wait_async0();
  __syncthreads();

  int cur = 0;
  for (int it = 1; it < KTOT / BK; ++it) {
    #pragma unroll
    for (int p = 0; p < 2; ++p) pa[p] += BK;
    #pragma unroll
    for (int p = 0; p < 4; ++p) pb[p] += BK;
    issue_tile(cur ^ 1);     // async DMA into the other buffer
    compute(cur);            // WMMAs run while the DMA streams in
    wait_async0();           // our async writes landed in LDS
    __syncthreads();         // everyone's writes visible
    cur ^= 1;
  }
  compute(cur);

  // ---- fused gate epilogue ----
  // C/D layout: VGPR r holds row M = hk*8 + r, column N = lm.
  #pragma unroll
  for (int ms = 0; ms < 2; ++ms) {
    #pragma unroll
    for (int ns = 0; ns < 2; ++ns) {
      const int hcol = colBase + waveN * 32 + ns * 16 + lm;
      float bias[4];
      #pragma unroll
      for (int g = 0; g < 4; ++g)
        bias[g] = bW[g * NH + hcol] + bU[g * NH + hcol];

      #pragma unroll
      for (int r = 0; r < 8; ++r) {
        const int b = rowBase + waveM * 32 + ms * 16 + hk * 8 + r;
        const float zf = acc[0][ms][ns][r] + bias[0];
        const float zi = acc[1][ms][ns][r] + bias[1];
        const float zo = acc[2][ms][ns][r] + bias[2];
        const float zg = acc[3][ms][ns][r] + bias[3];
        const float ft = fast_sigmoid(zf);
        const float it = fast_sigmoid(zi);
        const float ot = fast_sigmoid(zo);
        const float gt = fast_tanh(zg);
        const float cprev = cp[(size_t)b * NH + hcol];
        const float ct = ft * cprev + it * gt;
        const float ht = ot * fast_tanh(ct);
        out[(size_t)b * NH + hcol] = ht;
        out[(size_t)NB * NH + (size_t)b * NH + hcol] = ct;
      }
    }
  }
}

extern "C" void kernel_launch(void* const* d_in, const int* in_sizes, int n_in,
                              void* d_out, int out_size, void* d_ws, size_t ws_size,
                              hipStream_t stream) {
  (void)in_sizes; (void)n_in; (void)out_size; (void)ws_size;
  const float* x  = (const float*)d_in[0];
  const float* hp = (const float*)d_in[1];
  const float* cp = (const float*)d_in[2];
  const float* Wx = (const float*)d_in[3];
  const float* bW = (const float*)d_in[4];
  const float* Uh = (const float*)d_in[5];
  const float* bU = (const float*)d_in[6];
  float* out = (float*)d_out;

  // Workspace layout (48 MB): Ab bf16 8192x2048, then Wb bf16 4096x2048.
  __bf16* Ab = (__bf16*)d_ws;
  __bf16* Wb = Ab + (size_t)NB * KTOT;

  {
    const int groups = (NB + 4 * NH) * (KTOT / 8);   // 3,145,728
    pack_bf16<<<groups / 256, 256, 0, stream>>>(x, hp, Wx, Uh, Ab, Wb);
  }
  {
    dim3 grid(NH / 64, NB / 128, 1);                 // 16 x 64 blocks
    lstm_cell_wmma<<<grid, dim3(256), 0, stream>>>(Ab, Wb, cp, bW, bU, out);
  }
}
